// FeatMambaBlock_63720134803672
// MI455X (gfx1250) — compile-verified, matrix operand order checked by reading, not run on hardware
//
#include <hip/hip_runtime.h>
#include <cstdint>

// ---------------- CDNA5 WMMA / TDM types ----------------
typedef __attribute__((ext_vector_type(16))) _Float16     v16h;
typedef __attribute__((ext_vector_type(8)))  float        v8f;
typedef __attribute__((ext_vector_type(4)))  int          v4i;
typedef __attribute__((ext_vector_type(8)))  int          v8i;
typedef __attribute__((ext_vector_type(4)))  unsigned int v4u;

#define NL 32768          // L = 32*32*32 tokens per batch
#define NB 2              // batch
#define CH 64             // channels / d_inner
#define DSTATE 8
#define DTRANK 4

// K index inside a 16-half f16 fragment (ISA 7.12.2: 16-bit A/B layout).
// halves 0..7  -> K = hi*8 + e        (contiguous 16B chunk)
// halves 8..15 -> K = 16 + hi*8 + e-8 (contiguous 16B chunk)
__device__ __forceinline__ int kmap(int e, int hi) {
    return (e < 8) ? (hi * 8 + e) : (16 + hi * 8 + (e - 8));
}

// ---------------- TDM: load one contiguous 4KB row-tile into LDS ------------
// Copies 2048 f16 elements (32 rows x 64ch, c fastest) from global to LDS,
// inserting 4 DWORDs of padding after every 32 DWORDs so successive 128B rows
// land at a 144B LDS stride (bank-conflict-free fragment reads).
// D# per CDNA5 ISA ch.8: group0 = {count|flags, lds_addr, global_addr, type=2},
// group1 = {data_size=2B + pad cfg, dims/tiles/strides}.
// This toolchain exposes the 6-arg builtin (extra int32x8 before cpol).
__device__ __forceinline__ void tdm_load_row(uint32_t lds_off, uint64_t gaddr) {
    v4u g0;
    g0[0] = 1u;                                           // count=1 (user descriptor)
    g0[1] = lds_off;                                      // lds_addr (bytes)
    g0[2] = (uint32_t)(gaddr & 0xFFFFFFFFu);              // global_addr[31:0]
    g0[3] = (uint32_t)((gaddr >> 32) & 0x01FFFFFFu)       // global_addr[56:32]
          | 0x80000000u;                                  // type=2 ("image")
    v8i g1;
    g1[0] = (int)((1u << 16)      // data_size = 1 -> 2 bytes
                | (1u << 20)      // pad_enable
                | (4u << 22)      // pad_interval: code 4 -> every 32 DWORDs
                | (3u << 25));    // pad_amount: code 3 -> 4 DWORDs
    g1[1] = (int)(2048u << 16);   // tensor_dim0[15:0] = 2048 (atomic_addr = 0)
    g1[2] = (int)(1u << 16);      // tensor_dim0[31:16]=0 | tensor_dim1[15:0]=1
    g1[3] = (int)(2048u << 16);   // tensor_dim1[31:16]=0 | tile_dim0 = 2048
    g1[4] = 1;                    // tile_dim1 = 1, tile_dim2 = 0
    g1[5] = 2048;                 // tensor_dim0_stride[31:0]
    g1[6] = 0;
    g1[7] = 0;
    v4i z4 = {0, 0, 0, 0};
    v8i z8 = {0, 0, 0, 0, 0, 0, 0, 0};
    __builtin_amdgcn_tensor_load_to_lds(g0, g1, z4, z4, z8, 0);
}

// ---------------- weight packing: f32 matrix -> per-fragment f16 A layout ----
__global__ void pack_a_kernel(const float* __restrict__ W, _Float16* __restrict__ out,
                              int Mtiles, int Ksteps, int Mrows, int K, float scale) {
    int i = blockIdx.x * blockDim.x + threadIdx.x;
    int total = Mtiles * Ksteps * 512;
    if (i >= total) return;
    int e = i & 15, lane = (i >> 4) & 31, frag = i >> 9;
    int ks = frag % Ksteps, mt = frag / Ksteps;
    int m = mt * 16 + (lane & 15);
    int k = ks * 32 + kmap(e, lane >> 4);
    float v = (m < Mrows && k < K) ? W[m * K + k] * scale : 0.f;
    out[i] = (_Float16)v;
}

// conv weights (O=64, I=64, 3,3,3): fragment index = (t*2+ks)*4 + mt, t = tap 0..26
__global__ void pack_conv_kernel(const float* __restrict__ W, _Float16* __restrict__ out) {
    int i = blockIdx.x * blockDim.x + threadIdx.x;
    if (i >= 27 * 2 * 4 * 512) return;
    int e = i & 15, lane = (i >> 4) & 31, frag = i >> 9;
    int mt = frag & 3, ks = (frag >> 2) & 1, t = frag >> 3;
    int O = mt * 16 + (lane & 15);
    int I = ks * 32 + kmap(e, lane >> 4);
    out[i] = (_Float16)W[(O * 64 + I) * 27 + t];
}

// ---------------- convert channel-major f32 volume -> token-major f16 -------
// X[b][c][l] f32  ->  Y[b][l][c] f16   (l = h*1024 + w*32 + d)
__global__ void cvt_tokmajor_kernel(const float* __restrict__ X, _Float16* __restrict__ Y) {
    int i = blockIdx.x * 256 + threadIdx.x;
    if (i >= NB * CH * NL) return;
    int l = i & (NL - 1);
    int c = (i >> 15) & 63;
    int b = i >> 21;
    Y[((size_t)b * NL + l) * 64 + c] = (_Float16)X[((size_t)b * 64 + c) * NL + l];
}

// ---------------- conv3d (64->64, 3x3x3, pad 1) as implicit GEMM ------------
// Input X16 is token-major f16. Per block = one (b,h,w), full d-row of 32
// outputs. Staging: LDS zero-fill, then 9 TDM row-tile loads (one per valid
// (ih,iw) neighbor) with pad-stride 144B. 8 waves = 4 Mtiles x 2 Ntiles.
__global__ __launch_bounds__(256)
void conv3d_wmma_kernel(const _Float16* __restrict__ X16, const _Float16* __restrict__ Wpk,
                        const float* __restrict__ bias, const float* __restrict__ addend,
                        float* __restrict__ Y) {
    __shared__ _Float16 lx[9 * 36 * 72];   // [neigh][d_idx 0..35][ch], row stride 72 halves
    int b = blockIdx.x >> 10;
    int h = (blockIdx.x >> 5) & 31;
    int w = blockIdx.x & 31;
    int tid = threadIdx.x;

    // zero-fill LDS (handles d=-1/32 boundary rows and out-of-range neighbors)
    {
        v4i z4 = {0, 0, 0, 0};
        v4i* lp = (v4i*)lx;
        for (int i = tid; i < (9 * 36 * 72) / 8; i += 256) lp[i] = z4;
    }
    __syncthreads();

    // TDM staging: one descriptor per valid neighbor, issued from thread 0.
    if (tid == 0) {
        uint32_t lbase = (uint32_t)(uintptr_t)lx;   // flat LDS addr[31:0] == LDS offset
        for (int ng = 0; ng < 9; ++ng) {
            int ih = h + ng / 3 - 1, iw = w + ng % 3 - 1;
            if ((unsigned)ih < 32u && (unsigned)iw < 32u) {
                uint64_t ga = (uint64_t)(uintptr_t)
                    (X16 + (((size_t)b * 1024 + ih * 32 + iw) * 32) * 64);
                tdm_load_row(lbase + (uint32_t)(ng * 36 + 1) * 144u, ga);
            }
        }
    }
    __builtin_amdgcn_s_wait_tensorcnt(0);
    __syncthreads();

    int lane = tid & 31, wave = tid >> 5;
    int mt = wave & 3, nt = wave >> 2;
    int n0 = nt * 16;
    int hi = lane >> 4, nn = lane & 15;
    v8f acc = {0.f, 0.f, 0.f, 0.f, 0.f, 0.f, 0.f, 0.f};

    for (int ng = 0; ng < 9; ++ng) {
        for (int dd = 0; dd < 3; ++dd) {
            int t = ng * 3 + dd;
            int dbase = (ng * 36 + n0 + nn + dd) * 72;
            for (int ks = 0; ks < 2; ++ks) {
                v16h a = *reinterpret_cast<const v16h*>(
                    Wpk + (((t * 2 + ks) * 4 + mt) * 32 + lane) * 16);
                union { v16h hv; v4i q[2]; } u;
                int hb = dbase + ks * 32 + hi * 8;
                u.q[0] = *reinterpret_cast<const v4i*>(&lx[hb]);
                u.q[1] = *reinterpret_cast<const v4i*>(&lx[hb + 16]);
                acc = __builtin_amdgcn_wmma_f32_16x16x32_f16(
                    false, a, false, u.hv, (short)0, acc, false, false);
            }
        }
    }

    int d = n0 + nn;
    for (int r = 0; r < 8; ++r) {
        int O = mt * 16 + hi * 8 + r;
        size_t idx = (((size_t)b * 64 + O) * 1024 + h * 32 + w) * 32 + d;
        float v = acc[r] + bias[O];
        if (addend) v += addend[idx];
        Y[idx] = v;
    }
}

// ---------------- instance norm + relu over spatial per (b,c) ---------------
__global__ __launch_bounds__(256)
void instnorm_relu_kernel(const float* __restrict__ X, float* __restrict__ Y) {
    __shared__ float s1[256], s2[256];
    int bc = blockIdx.x;
    const float* p = X + (size_t)bc * NL;
    float sum = 0.f, sq = 0.f;
    for (int i = threadIdx.x; i < NL; i += 256) { float v = p[i]; sum += v; sq += v * v; }
    s1[threadIdx.x] = sum; s2[threadIdx.x] = sq; __syncthreads();
    for (int o = 128; o > 0; o >>= 1) {
        if (threadIdx.x < o) { s1[threadIdx.x] += s1[threadIdx.x + o]; s2[threadIdx.x] += s2[threadIdx.x + o]; }
        __syncthreads();
    }
    float m = s1[0] * (1.f / NL);
    float var = s2[0] * (1.f / NL) - m * m;
    float rs = rsqrtf(var + 1e-5f);
    float* q = Y + (size_t)bc * NL;
    for (int i = threadIdx.x; i < NL; i += 256) {
        float v = (p[i] - m) * rs;
        q[i] = v > 0.f ? v : 0.f;
    }
}

// ---------------- channel-major GEMM: Y[b,f,l] = sum_c A[f,c] X[b,c,l] ------
__global__ __launch_bounds__(256)
void gemm_cm_wmma_kernel(const float* __restrict__ X, const _Float16* __restrict__ Apk,
                         float* __restrict__ Y, int Mtiles) {
    __shared__ _Float16 lx[128 * 72];
    int l0 = blockIdx.x * 128;
    int b  = blockIdx.y;
    const float* Xb = X + (size_t)b * 64 * NL;
    for (int i = threadIdx.x; i < 64 * 128; i += 256) {
        int j = i & 127, c = i >> 7;
        lx[j * 72 + c] = (_Float16)Xb[(size_t)c * NL + l0 + j];
    }
    __syncthreads();

    int lane = threadIdx.x & 31, wave = threadIdx.x >> 5;
    int hi = lane >> 4, nn = lane & 15;
    for (int tp = wave; tp < Mtiles * 8; tp += 8) {
        int mt = tp >> 3, ns = tp & 7;
        v8f acc = {0.f, 0.f, 0.f, 0.f, 0.f, 0.f, 0.f, 0.f};
        for (int ks = 0; ks < 2; ++ks) {
            v16h a = *reinterpret_cast<const v16h*>(Apk + ((mt * 2 + ks) * 32 + lane) * 16);
            union { v16h hv; v4i q[2]; } u;
            int hb = (ns * 16 + nn) * 72 + ks * 32 + hi * 8;
            u.q[0] = *reinterpret_cast<const v4i*>(&lx[hb]);
            u.q[1] = *reinterpret_cast<const v4i*>(&lx[hb + 16]);
            acc = __builtin_amdgcn_wmma_f32_16x16x32_f16(
                false, a, false, u.hv, (short)0, acc, false, false);
        }
        int l = l0 + ns * 16 + nn;
        for (int r = 0; r < 8; ++r) {
            int f = mt * 16 + hi * 8 + r;
            Y[((size_t)b * Mtiles * 16 + f) * NL + l] = acc[r];
        }
    }
}

// ---------------- depthwise causal conv1d (window 4) + silu -----------------
__global__ void conv1d_silu_kernel(const float* __restrict__ xz, const float* __restrict__ w,
                                   const float* __restrict__ bias, float* __restrict__ xc, int dir) {
    int i = blockIdx.x * 256 + threadIdx.x;
    if (i >= NB * CH * NL) return;
    int l = i & (NL - 1);
    int d = (i >> 15) & 63;
    int b = i >> 21;
    const float* row = xz + ((size_t)b * 128 + d) * NL;  // x = rows 0..63 of xz
    float acc = bias[d];
#pragma unroll
    for (int j = 0; j < 4; ++j) {
        int li = l - 3 + j;
        if (li >= 0) {
            int src = dir ? (NL - 1 - li) : li;
            acc += w[d * 4 + j] * row[src];
        }
    }
    acc = acc / (1.f + __expf(-acc));
    xc[((size_t)b * CH + d) * NL + l] = acc;
}

// ---------------- selective scan: 8 state lanes per (b,d) -------------------
__global__ __launch_bounds__(256)
void scan_kernel(const float* __restrict__ xc, const float* __restrict__ dbl,
                 const float* __restrict__ xz, const float* __restrict__ dtw,
                 const float* __restrict__ dtb, const float* __restrict__ A_log,
                 const float* __restrict__ Dsk, float* __restrict__ ycomb, int dir) {
    int t = blockIdx.x * 256 + threadIdx.x;       // 1024 lanes total
    int n = t & 7;
    int d = (t >> 3) & 63;
    int b = t >> 9;
    float Av = -__expf(A_log[d * 8 + n]);
    float w0 = dtw[d * 4 + 0], w1 = dtw[d * 4 + 1], w2 = dtw[d * 4 + 2], w3 = dtw[d * 4 + 3];
    float db = dtb[d];
    float Dv = Dsk[d];
    const float* xcr = xc + ((size_t)b * CH + d) * NL;
    const float* r0 = dbl + ((size_t)b * 32 + 0) * NL;
    const float* r1 = dbl + ((size_t)b * 32 + 1) * NL;
    const float* r2 = dbl + ((size_t)b * 32 + 2) * NL;
    const float* r3 = dbl + ((size_t)b * 32 + 3) * NL;
    const float* Br = dbl + ((size_t)b * 32 + 4 + n) * NL;
    const float* Cr = dbl + ((size_t)b * 32 + 12 + n) * NL;
    const float* zr = xz + ((size_t)b * 128 + 64 + d) * NL;  // z = rows 64..127
    float* yr = ycomb + ((size_t)b * CH + d) * NL;

    float h = 0.f;
    for (int l = 0; l < NL; ++l) {
        float dt = w0 * r0[l] + w1 * r1[l] + w2 * r2[l] + w3 * r3[l] + db;
        dt = (dt > 20.f) ? dt : log1pf(__expf(dt));           // softplus
        float xv = xcr[l];
        h = __expf(dt * Av) * h + (dt * xv) * Br[l];
        float p = h * Cr[l];
        p += __shfl_xor(p, 1);
        p += __shfl_xor(p, 2);
        p += __shfl_xor(p, 4);
        if (n == 0) {
            int pos = dir ? (NL - 1 - l) : l;
            float z = zr[pos];
            float val = (p + xv * Dv) * (z / (1.f + __expf(-z)));
            if (dir) yr[pos] += val; else yr[pos] = val;
        }
    }
}

// ---------------- layernorm over 64 channels per token (in place) -----------
__global__ void layernorm_kernel(float* __restrict__ Y, const float* __restrict__ g,
                                 const float* __restrict__ bb) {
    int i = blockIdx.x * 256 + threadIdx.x;       // NB*NL columns
    int b = i >> 15, l = i & (NL - 1);
    float* col = Y + (size_t)b * CH * NL + l;
    float m = 0.f;
    for (int c = 0; c < 64; ++c) m += col[(size_t)c * NL];
    m *= (1.f / 64.f);
    float var = 0.f;
    for (int c = 0; c < 64; ++c) { float dv = col[(size_t)c * NL] - m; var += dv * dv; }
    var *= (1.f / 64.f);
    float rs = rsqrtf(var + 1e-5f);
    for (int c = 0; c < 64; ++c)
        col[(size_t)c * NL] = (col[(size_t)c * NL] - m) * rs * g[c] + bb[c];
}

// ---------------- host-side orchestration -----------------------------------
extern "C" void kernel_launch(void* const* d_in, const int* in_sizes, int n_in,
                              void* d_out, int out_size, void* d_ws, size_t ws_size,
                              hipStream_t stream) {
    const float* input   = (const float*)d_in[0];
    const float* conv_w  = (const float*)d_in[1];
    const float* conv_b  = (const float*)d_in[2];
    const float* in_proj = (const float*)d_in[3];
    const float* c1d_w   = (const float*)d_in[4];
    const float* c1d_b   = (const float*)d_in[5];
    const float* xproj   = (const float*)d_in[6];
    const float* dtw     = (const float*)d_in[7];
    const float* dtb     = (const float*)d_in[8];
    const float* A_log   = (const float*)d_in[9];
    const float* Ab_log  = (const float*)d_in[10];
    const float* Dsk     = (const float*)d_in[11];
    const float* outproj = (const float*)d_in[12];
    const float* ln_g    = (const float*)d_in[13];
    const float* ln_b    = (const float*)d_in[14];
    float* ws  = (float*)d_ws;
    float* out = (float*)d_out;

    const size_t SZV = (size_t)NB * CH * NL;      // one (b,64,L) volume in floats
    float* skip  = ws;                            // relu(instnorm(conv1))
    float* conv1 = ws + SZV;                      // conv#1 out, reused for out_proj/LN
    float* xz    = ws + 2 * SZV;                  // (b,128,L)
    float* xcf   = ws + 4 * SZV;
    float* xcb   = ws + 5 * SZV;
    float* dblf  = ws + 6 * SZV;                  // (b,32,L)
    float* dblb  = dblf + SZV / 2;
    float* ycomb = ws + 7 * SZV;
    _Float16* wpk_conv = (_Float16*)(ws + 8 * SZV);      // packed weights (<=125K halves)
    _Float16* wpk_in   = wpk_conv + 110592;
    _Float16* wpk_x    = wpk_in + 8 * 2 * 512;
    _Float16* wpk_out  = wpk_x + 2 * 2 * 512;
    _Float16* x16a = (_Float16*)(ws + 8 * SZV + 65536);  // token-major f16 conv inputs
    _Float16* x16b = x16a + (size_t)NB * NL * 64;

    // pack weights into WMMA fragment layout (f16); fold /2 into out_proj
    pack_conv_kernel<<<(110592 + 255) / 256, 256, 0, stream>>>(conv_w, wpk_conv);
    pack_a_kernel<<<(8 * 2 * 512 + 255) / 256, 256, 0, stream>>>(in_proj, wpk_in, 8, 2, 128, 64, 1.f);
    pack_a_kernel<<<(2 * 2 * 512 + 255) / 256, 256, 0, stream>>>(xproj, wpk_x, 2, 2, 20, 64, 1.f);
    pack_a_kernel<<<(4 * 2 * 512 + 255) / 256, 256, 0, stream>>>(outproj, wpk_out, 4, 2, 64, 64, 0.5f);

    // skip branch (conv input staged via TDM from token-major f16 mirror)
    cvt_tokmajor_kernel<<<(NB * CH * NL) / 256, 256, 0, stream>>>(input, x16a);
    conv3d_wmma_kernel<<<2048, 256, 0, stream>>>(x16a, wpk_conv, conv_b, nullptr, conv1);
    instnorm_relu_kernel<<<NB * CH, 256, 0, stream>>>(conv1, skip);

    // mamba branch
    gemm_cm_wmma_kernel<<<dim3(NL / 128, NB), 256, 0, stream>>>(input, wpk_in, xz, 8);
    conv1d_silu_kernel<<<(NB * CH * NL) / 256, 256, 0, stream>>>(xz, c1d_w, c1d_b, xcf, 0);
    conv1d_silu_kernel<<<(NB * CH * NL) / 256, 256, 0, stream>>>(xz, c1d_w, c1d_b, xcb, 1);
    gemm_cm_wmma_kernel<<<dim3(NL / 128, NB), 256, 0, stream>>>(xcf, wpk_x, dblf, 2);
    gemm_cm_wmma_kernel<<<dim3(NL / 128, NB), 256, 0, stream>>>(xcb, wpk_x, dblb, 2);
    scan_kernel<<<4, 256, 0, stream>>>(xcf, dblf, xz, dtw, dtb, A_log, Dsk, ycomb, 0);
    scan_kernel<<<4, 256, 0, stream>>>(xcb, dblb, xz, dtw, dtb, Ab_log, Dsk, ycomb, 1);

    // out projection (+/2 folded), layernorm, final conv + skip
    gemm_cm_wmma_kernel<<<dim3(NL / 128, NB), 256, 0, stream>>>(ycomb, wpk_out, conv1, 4);
    layernorm_kernel<<<(NB * NL) / 256, 256, 0, stream>>>(conv1, ln_g, ln_b);
    cvt_tokmajor_kernel<<<(NB * CH * NL) / 256, 256, 0, stream>>>(conv1, x16b);
    conv3d_wmma_kernel<<<2048, 256, 0, stream>>>(x16b, wpk_conv, conv_b, skip, out);
}
